// EventSplitter_40458591929202
// MI455X (gfx1250) — compile-verified, matrix operand order checked by reading, not run on hardware
//
#include <hip/hip_runtime.h>
#include <hip/hip_bf16.h>

// ---------------------------------------------------------------------------
// EventSplitter graph-transformer forward for MI455X (gfx1250, wave32, WMMA).
// Dense GEMMs: v_wmma_f32_16x16x32_f16 with weights pre-permuted into the
// WMMA B-fragment layout (per-lane contiguous -> ds_load_b128 x2 per frag).
// A-tile staging is fully vectorized: 2x global_load_b128 -> 1x ds_store_b128.
// Irregular segment-softmax attention uses VALU + global float atomics.
// ---------------------------------------------------------------------------

typedef __attribute__((ext_vector_type(16))) _Float16 v16h;
typedef __attribute__((ext_vector_type(8)))  float    v8f;

#define HID    192
#define HEADS  4
#define DH     48
#define FFN    384
#define NLAYER 3
#define QKVW   (3 * HID)   // fused q|k|v width = 576

// ---------------- WMMA fragment K index (ISA 7.12.2) -----------------------
// lanes 0-15: halves 0..7 -> K 0..7,  halves 8..15 -> K 16..23
// lanes 16-31: halves 0..7 -> K 8..15, halves 8..15 -> K 24..31
__device__ __host__ inline int frag_k(int lane, int h) {
    return ((lane >> 4) << 3) + ((h < 8) ? h : (h + 8));
}

// A (16x32 f16, MxK) from row-major LDS tile: per-lane halves are two
// contiguous 16B runs -> compiler emits 2x ds_load_b128.
__device__ inline v16h load_frag_a(const _Float16* __restrict__ s, int stride) {
    int lane = threadIdx.x & 31;
    int m = lane & 15;
    v16h r;
#pragma unroll
    for (int h = 0; h < 16; ++h) r[h] = s[m * stride + frag_k(lane, h)];
    return r;
}

// B fragment from pre-permuted LDS: lane's 16 halves are fully contiguous.
__device__ inline v16h load_frag_b(const _Float16* __restrict__ Bs, int ct) {
    int lane = threadIdx.x & 31;
    return *(const v16h*)(Bs + (((size_t)(ct * 32 + lane)) << 4));
}

// ---------------- generic tiled WMMA GEMM ----------------------------------
// C[M x ncr] = act( A[M x Kpad] * Wperm + bias ), f32 out.
// Wperm layout: [Kpad/32][NCOLS/16][32 lanes][16 halves] (B-fragment order).
// MODE: 0 = bias only, 1 = bias + ReLU.
// PAIR: A row g gathered on the fly: [h[src[g]], h[dst[g]], eattr[g], 0pad].
template <int NCOLS, int BR, int MODE, bool PAIR>
__global__ __launch_bounds__(256) void gemm_tile(
    const float* __restrict__ A, const _Float16* __restrict__ Bh,
    const float* __restrict__ bias, float* __restrict__ C,
    int M, int Kpad, int ncr,
    const float* __restrict__ hgat, const int* __restrict__ esrc,
    const int* __restrict__ edst, const float* __restrict__ eattr)
{
    constexpr int CT     = NCOLS / 16;        // column tiles
    constexpr int STRIPS = BR / 16;           // row strips per block
    constexpr int TILES  = STRIPS * CT;
    constexpr int FRAGS  = (TILES + 7) / 8;
    constexpr int ASTR   = 40;                // padded LDS stride for A tile (80B rows)

    __shared__ alignas(16) _Float16 As[BR * ASTR];
    __shared__ alignas(32) _Float16 Bs[32 * NCOLS];  // CT * 512 halves

    const int tid  = threadIdx.x;
    const int wave = tid >> 5;
    const int lane = tid & 31;
    const long long rowBase = (long long)blockIdx.x * BR;

    v8f acc[FRAGS];
#pragma unroll
    for (int f = 0; f < FRAGS; ++f)
#pragma unroll
        for (int i = 0; i < 8; ++i) acc[f][i] = 0.0f;

    for (int k0 = 0; k0 < Kpad; k0 += 32) {
        // --- stage A tile (f32 -> f16), 8-element runs per thread ---------
        // run r covers row (run>>2), cols (run&3)*8 .. +7 of the 32-wide tile.
        for (int run = tid; run < BR * 4; run += 256) {
            int r  = run >> 2;
            int c0 = (run & 3) * 8;
            long long grow = rowBase + r;
            union { _Float16 hv[8]; uint4 u; } pk;
            if (grow < M) {
                int k = k0 + c0;
                if constexpr (PAIR) {
                    if (k + 8 <= HID) {                       // fully in h[src]
                        const float* ap = hgat + (size_t)esrc[grow] * HID + k;
                        float4 f0 = *(const float4*)ap;
                        float4 f1 = *(const float4*)(ap + 4);
                        pk.hv[0] = (_Float16)f0.x; pk.hv[1] = (_Float16)f0.y;
                        pk.hv[2] = (_Float16)f0.z; pk.hv[3] = (_Float16)f0.w;
                        pk.hv[4] = (_Float16)f1.x; pk.hv[5] = (_Float16)f1.y;
                        pk.hv[6] = (_Float16)f1.z; pk.hv[7] = (_Float16)f1.w;
                    } else if (k >= HID && k + 8 <= 2 * HID) { // fully in h[dst]
                        const float* ap = hgat + (size_t)edst[grow] * HID + (k - HID);
                        float4 f0 = *(const float4*)ap;
                        float4 f1 = *(const float4*)(ap + 4);
                        pk.hv[0] = (_Float16)f0.x; pk.hv[1] = (_Float16)f0.y;
                        pk.hv[2] = (_Float16)f0.z; pk.hv[3] = (_Float16)f0.w;
                        pk.hv[4] = (_Float16)f1.x; pk.hv[5] = (_Float16)f1.y;
                        pk.hv[6] = (_Float16)f1.z; pk.hv[7] = (_Float16)f1.w;
                    } else {                                   // eattr + zero pad tail
#pragma unroll
                        for (int j = 0; j < 8; ++j) {
                            int kk = k + j;
                            float v = 0.0f;
                            if (kk >= 2 * HID && kk < 2 * HID + 4)
                                v = eattr[(size_t)grow * 4 + (kk - 2 * HID)];
                            pk.hv[j] = (_Float16)v;
                        }
                    }
                } else {
                    const float* ap = A + (size_t)grow * Kpad + k;
                    float4 f0 = *(const float4*)ap;
                    float4 f1 = *(const float4*)(ap + 4);
                    pk.hv[0] = (_Float16)f0.x; pk.hv[1] = (_Float16)f0.y;
                    pk.hv[2] = (_Float16)f0.z; pk.hv[3] = (_Float16)f0.w;
                    pk.hv[4] = (_Float16)f1.x; pk.hv[5] = (_Float16)f1.y;
                    pk.hv[6] = (_Float16)f1.z; pk.hv[7] = (_Float16)f1.w;
                }
            } else {
#pragma unroll
                for (int j = 0; j < 8; ++j) pk.hv[j] = (_Float16)0.0f;
            }
            *(uint4*)&As[r * ASTR + c0] = pk.u;   // 16B-aligned: r*80 + c0*2
        }
        // --- stage B k-block: straight coalesced copy (already permuted) ---
        {
            const uint4* gB = (const uint4*)(Bh + ((size_t)(k0 >> 5)) * CT * 512);
            uint4* sB = (uint4*)Bs;
            for (int i = tid; i < CT * 64; i += 256) sB[i] = gB[i];
        }
        __syncthreads();

#pragma unroll
        for (int f = 0; f < FRAGS; ++f) {
            int t = wave + f * 8;
            if (t < TILES) {
                int strip = t / CT, ct = t % CT;
                v16h af = load_frag_a(&As[strip * 16 * ASTR], ASTR);
                v16h bf = load_frag_b(Bs, ct);
                acc[f] = __builtin_amdgcn_wmma_f32_16x16x32_f16(
                    false, af, false, bf, (short)0, acc[f], false, false);
            }
        }
        __syncthreads();
    }

    // --- store: C frag layout: VGPR r -> M = r + (lane<16?0:8), N = lane&15 ---
    const int n  = lane & 15;
    const int mb = (lane >> 4) << 3;
#pragma unroll
    for (int f = 0; f < FRAGS; ++f) {
        int t = wave + f * 8;
        if (t >= TILES) continue;
        int strip = t / CT, ct = t % CT;
        int col = ct * 16 + n;
        if (col >= ncr) continue;
        float bv = bias ? bias[col] : 0.0f;
#pragma unroll
        for (int r = 0; r < 8; ++r) {
            long long row = rowBase + strip * 16 + mb + r;
            if (row < M) {
                float v = acc[f][r] + bv;
                if (MODE == 1) v = fmaxf(v, 0.0f);
                C[(size_t)row * ncr + col] = v;
            }
        }
    }
}

// ---------------- weight f32 -> f16, permuted to B-fragment order ----------
// Writes columns [colOff, colOff+Nreal) of a CTtot*16-wide permuted buffer:
// Wh[((kb*CTtot + ct)*32 + lane)*16 + h] = W[kb*32 + frag_k(lane,h)][col-colOff]
__global__ void convert_w_kernel(const float* __restrict__ W, _Float16* __restrict__ Wh,
                                 int Kreal, int Kpad, int Nreal, int colOff, int CTtot)
{
    int ctCount = (Nreal + 15) / 16;
    int KB = Kpad >> 5;
    long long i = (long long)blockIdx.x * blockDim.x + threadIdx.x;
    long long tot = (long long)KB * ctCount * 512;
    if (i >= tot) return;
    int h    = (int)(i & 15);
    int lane = (int)((i >> 4) & 31);
    long long rest = i >> 9;
    int ctLocal = (int)(rest % ctCount);
    int kb      = (int)(rest / ctCount);
    int ct  = (colOff >> 4) + ctLocal;
    int k   = kb * 32 + frag_k(lane, h);
    int col = ct * 16 + (lane & 15);
    int wc  = col - colOff;
    float v = 0.0f;
    if (k < Kreal && wc >= 0 && wc < Nreal) v = W[(size_t)k * Nreal + wc];
    Wh[(((size_t)kb * CTtot + ct) * 32 + lane) * 16 + h] = (_Float16)v;
}

// ---------------- feature assembly + input projection ----------------------
__global__ void feat_in_kernel(const float* __restrict__ x, const int* __restrict__ batch,
                               const int* __restrict__ gptr, const int* __restrict__ tgid,
                               const float* __restrict__ gprob, const float* __restrict__ sprob,
                               const float* __restrict__ eppred,
                               const float* __restrict__ Win, const float* __restrict__ bin,
                               float* __restrict__ h, int N, int G)
{
    int node = blockIdx.x;
    int c = threadIdx.x;
    if (node >= N || c >= HID) return;
    int gid = gptr[batch[node]] + tgid[node];
    gid = gid < 0 ? 0 : (gid > G - 1 ? G - 1 : gid);
    float feat[16];
#pragma unroll
    for (int i = 0; i < 4; ++i) feat[i]      = x[(size_t)node * 4 + i];
#pragma unroll
    for (int i = 0; i < 3; ++i) feat[4 + i]  = gprob[(size_t)gid * 3 + i];
#pragma unroll
    for (int i = 0; i < 3; ++i) feat[7 + i]  = sprob[(size_t)node * 3 + i];
#pragma unroll
    for (int i = 0; i < 6; ++i) feat[10 + i] = eppred[(size_t)gid * 6 + i];
    float acc = bin[c];
#pragma unroll
    for (int k = 0; k < 16; ++k) acc += feat[k] * Win[k * HID + c];
    h[(size_t)node * HID + c] = acc;
}

// ---------------- ordered-uint encoding for float atomicMax ----------------
__device__ inline unsigned ord_enc(float f) {
    unsigned u = __float_as_uint(f);
    return (u & 0x80000000u) ? ~u : (u | 0x80000000u);
}
__device__ inline float ord_dec(unsigned u) {
    unsigned v = (u & 0x80000000u) ? (u & 0x7FFFFFFFu) : ~u;
    return __uint_as_float(v);
}

// ---------------- edge attention: logits + segment max ---------------------
__global__ void edge_logit_kernel(const float* __restrict__ q, const float* __restrict__ k,
                                  int ld, const float* __restrict__ We,
                                  const float* __restrict__ eattr,
                                  const int* __restrict__ src, const int* __restrict__ dst,
                                  float* __restrict__ logit, unsigned* __restrict__ mbits, int E)
{
    long long i = (long long)blockIdx.x * blockDim.x + threadIdx.x;
    if (i >= (long long)E * HEADS) return;
    int e = (int)(i >> 2), hh = (int)(i & 3);
    int s = src[e], d = dst[e];
    float ea0 = eattr[(size_t)e * 4 + 0], ea1 = eattr[(size_t)e * 4 + 1];
    float ea2 = eattr[(size_t)e * 4 + 2], ea3 = eattr[(size_t)e * 4 + 3];
    int base = hh * DH;
    float acc = 0.0f;
#pragma unroll 8
    for (int j = 0; j < DH; ++j) {
        int c = base + j;
        float ev = ea0 * We[c] + ea1 * We[HID + c] + ea2 * We[2 * HID + c] + ea3 * We[3 * HID + c];
        acc += q[(size_t)d * ld + c] * (k[(size_t)s * ld + c] + ev);
    }
    acc *= 0.14433756729740645f; // 1/sqrt(48)
    logit[i] = acc;
    atomicMax(&mbits[(size_t)d * HEADS + hh], ord_enc(acc));
}

// ---------------- edge attention: exp + segment sum ------------------------
__global__ void edge_exp_kernel(const float* __restrict__ logit, const unsigned* __restrict__ mbits,
                                const int* __restrict__ dst, float* __restrict__ aexp,
                                float* __restrict__ den, int E)
{
    long long i = (long long)blockIdx.x * blockDim.x + threadIdx.x;
    if (i >= (long long)E * HEADS) return;
    int e = (int)(i >> 2), hh = (int)(i & 3);
    int d = dst[e];
    float m = ord_dec(mbits[(size_t)d * HEADS + hh]);
    if (!(fabsf(m) <= 3.4e38f)) m = 0.0f;  // NaN / inf guard (empty segments)
    float a = expf(logit[i] - m);
    aexp[i] = a;
    atomicAdd(&den[(size_t)d * HEADS + hh], a);
}

// ---------------- edge attention: weighted message scatter -----------------
__global__ void edge_msg_kernel(const float* __restrict__ aexp, const float* __restrict__ den,
                                const float* __restrict__ v, int ld,
                                const float* __restrict__ We,
                                const float* __restrict__ eattr, const int* __restrict__ src,
                                const int* __restrict__ dst, float* __restrict__ msg, int E)
{
    long long i = (long long)blockIdx.x * blockDim.x + threadIdx.x;
    if (i >= (long long)E * HEADS) return;
    int e = (int)(i >> 2), hh = (int)(i & 3);
    int s = src[e], d = dst[e];
    float a = aexp[i] / (den[(size_t)d * HEADS + hh] + 1e-16f);
    float ea0 = eattr[(size_t)e * 4 + 0], ea1 = eattr[(size_t)e * 4 + 1];
    float ea2 = eattr[(size_t)e * 4 + 2], ea3 = eattr[(size_t)e * 4 + 3];
    int base = hh * DH;
#pragma unroll 8
    for (int j = 0; j < DH; ++j) {
        int c = base + j;
        float ev = ea0 * We[c] + ea1 * We[HID + c] + ea2 * We[2 * HID + c] + ea3 * We[3 * HID + c];
        atomicAdd(&msg[(size_t)d * HID + c], a * (v[(size_t)s * ld + c] + ev));
    }
}

// ---------------- residual + LayerNorm (in place on h) ---------------------
__global__ void ln_kernel(float* __restrict__ hbuf, const float* __restrict__ delta,
                          const float* __restrict__ g, const float* __restrict__ b, int N)
{
    __shared__ float red[256];
    int node = blockIdx.x;
    int c = threadIdx.x;
    float v = 0.0f;
    if (c < HID) v = hbuf[(size_t)node * HID + c] + delta[(size_t)node * HID + c];
    red[c] = v;
    __syncthreads();
    for (int s = 128; s > 0; s >>= 1) { if (c < s) red[c] += red[c + s]; __syncthreads(); }
    float mean = red[0] * (1.0f / HID);
    __syncthreads();
    float d = (c < HID) ? (v - mean) : 0.0f;
    red[c] = d * d;
    __syncthreads();
    for (int s = 128; s > 0; s >>= 1) { if (c < s) red[c] += red[c + s]; __syncthreads(); }
    float var = red[0] * (1.0f / HID);
    if (c < HID)
        hbuf[(size_t)node * HID + c] = d * rsqrtf(var + 1e-5f) * g[c] + b[c];
}

// ---------------------------------------------------------------------------
extern "C" void kernel_launch(void* const* d_in, const int* in_sizes, int n_in,
                              void* d_out, int out_size, void* d_ws, size_t ws_size,
                              hipStream_t stream)
{
    const float* x      = (const float*)d_in[0];
    const int*   eidx   = (const int*)  d_in[1];
    const float* eattr  = (const float*)d_in[2];
    const int*   batch  = (const int*)  d_in[3];
    const int*   gptr   = (const int*)  d_in[4];
    const int*   tgid   = (const int*)  d_in[5];
    const float* gprob  = (const float*)d_in[6];
    const float* sprob  = (const float*)d_in[7];
    const float* eppred = (const float*)d_in[8];
    const float* W_in   = (const float*)d_in[9];
    const float* b_in   = (const float*)d_in[10];
    const float* Wq     = (const float*)d_in[11];
    const float* Wk     = (const float*)d_in[12];
    const float* Wv     = (const float*)d_in[13];
    const float* We     = (const float*)d_in[14];
    const float* Wo     = (const float*)d_in[15];
    const float* bo     = (const float*)d_in[16];
    const float* ln1g   = (const float*)d_in[17];
    const float* ln1b   = (const float*)d_in[18];
    const float* Wff1   = (const float*)d_in[19];
    const float* bff1   = (const float*)d_in[20];
    const float* Wff2   = (const float*)d_in[21];
    const float* bff2   = (const float*)d_in[22];
    const float* ln2g   = (const float*)d_in[23];
    const float* ln2b   = (const float*)d_in[24];
    const float* We1    = (const float*)d_in[25];
    const float* be1    = (const float*)d_in[26];
    const float* We2    = (const float*)d_in[27];
    const float* be2    = (const float*)d_in[28];
    const float* We3    = (const float*)d_in[29];
    const float* be3    = (const float*)d_in[30];

    const int N = in_sizes[0] / 4;
    const int E = in_sizes[1] / 2;
    const int G = in_sizes[6] / 3;
    const int* src = eidx;
    const int* dst = eidx + E;

    // ---- workspace bump allocator ----
    char* p = (char*)d_ws;
    auto alloc = [&](size_t bytes) -> void* {
        void* r = (void*)p;
        p += (bytes + 255) & ~(size_t)255;
        return r;
    };
    float*    h     = (float*)   alloc((size_t)N * HID * 4);
    float*    qkv   = (float*)   alloc((size_t)N * QKVW * 4);  // fused q|k|v
    float*    msg   = (float*)   alloc((size_t)N * HID * 4);   // attn msg, also ff2 out
    float*    tmp   = (float*)   alloc((size_t)N * FFN * 4);   // msg@Wo out, ff1 out
    float*    logit = (float*)   alloc((size_t)E * HEADS * 4);
    float*    aexp  = (float*)   alloc((size_t)E * HEADS * 4);
    unsigned* mbits = (unsigned*)alloc((size_t)N * HEADS * 4);
    float*    den   = (float*)   alloc((size_t)N * HEADS * 4);
    _Float16* wh    = (_Float16*)alloc((size_t)131072 * 2);    // one permuted f16 weight at a time
    float*    z1    = (float*)   alloc((size_t)E * HID * 4);
    float*    z2    = (float*)   alloc((size_t)E * (HID / 2) * 4);

    const long long EH = (long long)E * HEADS;
    const int edgeGrid = (int)((EH + 255) / 256);

    auto conv = [&](const float* W, int Kr, int Kp, int Nr, int colOff, int CTtot) {
        long long tot = (long long)(Kp >> 5) * ((Nr + 15) / 16) * 512;
        convert_w_kernel<<<(int)((tot + 255) / 256), 256, 0, stream>>>(
            W, wh, Kr, Kp, Nr, colOff, CTtot);
    };

    // ---- input projection ----
    feat_in_kernel<<<N, HID, 0, stream>>>(x, batch, gptr, tgid, gprob, sprob, eppred,
                                          W_in, b_in, h, N, G);

    // ---- transformer layers ----
    for (int l = 0; l < NLAYER; ++l) {
        const float* Wel = We + (size_t)l * 4 * HID;

        // fused QKV: [Wq|Wk|Wv] permuted into one 576-wide buffer
        conv(Wq + (size_t)l * HID * HID, HID, HID, HID, 0,       QKVW / 16);
        conv(Wk + (size_t)l * HID * HID, HID, HID, HID, HID,     QKVW / 16);
        conv(Wv + (size_t)l * HID * HID, HID, HID, HID, 2 * HID, QKVW / 16);
        gemm_tile<QKVW, 32, 0, false><<<(N + 31) / 32, 256, 0, stream>>>(
            h, wh, nullptr, qkv, N, HID, QKVW, nullptr, nullptr, nullptr, nullptr);

        hipMemsetAsync(mbits, 0, (size_t)N * HEADS * 4, stream);
        hipMemsetAsync(den,   0, (size_t)N * HEADS * 4, stream);
        hipMemsetAsync(msg,   0, (size_t)N * HID * 4,   stream);

        edge_logit_kernel<<<edgeGrid, 256, 0, stream>>>(
            qkv, qkv + HID, QKVW, Wel, eattr, src, dst, logit, mbits, E);
        edge_exp_kernel<<<edgeGrid, 256, 0, stream>>>(logit, mbits, dst, aexp, den, E);
        edge_msg_kernel<<<edgeGrid, 256, 0, stream>>>(
            aexp, den, qkv + 2 * HID, QKVW, Wel, eattr, src, dst, msg, E);

        conv(Wo + (size_t)l * HID * HID, HID, HID, HID, 0, HID / 16);
        gemm_tile<HID, 64, 0, false><<<(N + 63) / 64, 256, 0, stream>>>(
            msg, wh, bo + (size_t)l * HID, tmp, N, HID, HID,
            nullptr, nullptr, nullptr, nullptr);
        ln_kernel<<<N, 256, 0, stream>>>(h, tmp, ln1g + (size_t)l * HID, ln1b + (size_t)l * HID, N);

        conv(Wff1 + (size_t)l * HID * FFN, HID, HID, FFN, 0, FFN / 16);
        gemm_tile<FFN, 64, 1, false><<<(N + 63) / 64, 256, 0, stream>>>(
            h, wh, bff1 + (size_t)l * FFN, tmp, N, HID, FFN,
            nullptr, nullptr, nullptr, nullptr);
        conv(Wff2 + (size_t)l * FFN * HID, FFN, FFN, HID, 0, HID / 16);
        gemm_tile<HID, 64, 0, false><<<(N + 63) / 64, 256, 0, stream>>>(
            tmp, wh, bff2 + (size_t)l * HID, msg, N, FFN, HID,
            nullptr, nullptr, nullptr, nullptr);
        ln_kernel<<<N, 256, 0, stream>>>(h, msg, ln2g + (size_t)l * HID, ln2b + (size_t)l * HID, N);
    }

    // ---- edge head ----
    // layer 1: K = 2*HID + 4 = 388, padded to 416; pair rows gathered in-kernel.
    conv(We1, 2 * HID + 4, 416, HID, 0, HID / 16);
    gemm_tile<HID, 64, 1, true><<<(E + 63) / 64, 256, 0, stream>>>(
        nullptr, wh, be1, z1, E, 416, HID, h, src, dst, eattr);
    // layer 2: 192 -> 96, ReLU
    conv(We2, HID, HID, HID / 2, 0, (HID / 2) / 16);
    gemm_tile<HID / 2, 64, 1, false><<<(E + 63) / 64, 256, 0, stream>>>(
        z1, wh, be2, z2, E, HID, HID / 2, nullptr, nullptr, nullptr, nullptr);
    // layer 3: 96 -> 1 (N padded to 16), write d_out directly
    conv(We3, HID / 2, HID / 2, 1, 0, 1);
    gemm_tile<16, 64, 0, false><<<(E + 63) / 64, 256, 0, stream>>>(
        z2, wh, be3, (float*)d_out, E, HID / 2, 1, nullptr, nullptr, nullptr, nullptr);
}
